// EncoderLayer_6682969113189
// MI455X (gfx1250) — compile-verified, matrix operand order checked by reading, not run on hardware
//
#include <hip/hip_runtime.h>
#include <hip/hip_bf16.h>

#define DIMC 256
#define EDIMC 32
#define HEADSC 8
#define DFFC 1024

typedef __bf16 bf16x16 __attribute__((ext_vector_type(16)));
typedef float  f32x8   __attribute__((ext_vector_type(8)));

union ABFrag { bf16x16 v; unsigned u[8]; };

static __device__ inline unsigned short f2bf(float f) {
    unsigned u = __float_as_uint(f);
    u += 0x7fffu + ((u >> 16) & 1u);   // round-to-nearest-even
    return (unsigned short)(u >> 16);
}

static __device__ inline unsigned fkey(float f) {
    unsigned u = __float_as_uint(f);
    return (u & 0x80000000u) ? ~u : (u | 0x80000000u);
}
static __device__ inline float fkeyinv(unsigned k) {
    return (k & 0x80000000u) ? __uint_as_float(k ^ 0x80000000u) : __uint_as_float(~k);
}

// ---------------------------------------------------------------------------
// Tiled GEMM: C[M,Nc] = A[M,K] @ B[K,Nc] (+bias) (optional ReLU)
// fp32 in / fp32 out, bf16 WMMA with f32 accumulation.
// Block = 128 threads (4 waves), tile 64x64, K-step 32.
// ---------------------------------------------------------------------------
__global__ __launch_bounds__(128)
void gemm_wmma_bf16(const float* __restrict__ A, const float* __restrict__ B,
                    const float* __restrict__ bias, float* __restrict__ C,
                    int M, int Nc, int K, int relu) {
    __shared__ __attribute__((aligned(16))) unsigned short lA[64][34]; // [m][k]
    __shared__ __attribute__((aligned(16))) unsigned short lB[64][34]; // [n][k] (B transposed)

    const int tid  = threadIdx.x;
    const int lane = tid & 31;
    const int wave = tid >> 5;
    const int lm   = lane & 15;
    const int half = lane >> 4;
    const int bm   = blockIdx.x * 64;
    const int bn   = blockIdx.y * 64;

    f32x8 acc[4] = {};

    for (int kt = 0; kt < K; kt += 32) {
        // ---- stage A tile (64 rows x 32 k), convert to bf16 ----
        {
            int row = tid >> 1;
            int cb  = (tid & 1) * 16;
            int gr  = bm + row;
            float4 v0, v1, v2, v3;
            if (gr < M) {
                const float4* p = (const float4*)(A + (size_t)gr * K + kt + cb);
                v0 = p[0]; v1 = p[1]; v2 = p[2]; v3 = p[3];
            } else {
                v0 = v1 = v2 = v3 = make_float4(0.f, 0.f, 0.f, 0.f);
            }
            float tmp[16] = {v0.x,v0.y,v0.z,v0.w, v1.x,v1.y,v1.z,v1.w,
                             v2.x,v2.y,v2.z,v2.w, v3.x,v3.y,v3.z,v3.w};
            #pragma unroll
            for (int i = 0; i < 16; ++i) lA[row][cb + i] = f2bf(tmp[i]);
        }
        // ---- stage B tile (32 k x 64 n) transposed to [n][k] ----
        {
            int k  = tid >> 2;
            int nb = (tid & 3) * 16;
            const float4* p = (const float4*)(B + (size_t)(kt + k) * Nc + bn + nb);
            float4 v0 = p[0], v1 = p[1], v2 = p[2], v3 = p[3];
            float tmp[16] = {v0.x,v0.y,v0.z,v0.w, v1.x,v1.y,v1.z,v1.w,
                             v2.x,v2.y,v2.z,v2.w, v3.x,v3.y,v3.z,v3.w};
            #pragma unroll
            for (int i = 0; i < 16; ++i) lB[nb + i][k] = f2bf(tmp[i]);
        }
        __syncthreads();

        // ---- A fragment for this wave's 16 rows ----
        ABFrag a;
        {
            int ar = wave * 16 + lm;
            #pragma unroll
            for (int i = 0; i < 8; ++i) {
                int kk = ((i < 4) ? 0 : 16) + half * 8 + (i & 3) * 2;
                a.u[i] = *(const unsigned*)&lA[ar][kk];
            }
        }
        // ---- 4 N-fragments, 4 WMMAs ----
        #pragma unroll
        for (int f = 0; f < 4; ++f) {
            ABFrag b;
            int bc = f * 16 + lm;
            #pragma unroll
            for (int i = 0; i < 8; ++i) {
                int kk = ((i < 4) ? 0 : 16) + half * 8 + (i & 3) * 2;
                b.u[i] = *(const unsigned*)&lB[bc][kk];
            }
            acc[f] = __builtin_amdgcn_wmma_f32_16x16x32_bf16(
                false, a.v, false, b.v, (short)0, acc[f], false, false);
        }
        __syncthreads();
    }

    // ---- epilogue ----
    #pragma unroll
    for (int f = 0; f < 4; ++f) {
        int col = bn + f * 16 + lm;
        float bv = bias ? bias[col] : 0.f;
        #pragma unroll
        for (int r = 0; r < 8; ++r) {
            int m = bm + wave * 16 + r + half * 8;
            if (m < M) {
                float v = acc[f][r] + bv;
                if (relu) v = v > 0.f ? v : 0.f;
                C[(size_t)m * Nc + col] = v;
            }
        }
    }
}

// ---------------------------------------------------------------------------
// Self-loop edge-attr mean: segment-sum ew by dst + count
// ---------------------------------------------------------------------------
__global__ void loopea_accum(const int* __restrict__ dst, const float* __restrict__ ew,
                             float* __restrict__ loopea, int* __restrict__ cnt, int E) {
    int g = blockIdx.x * blockDim.x + threadIdx.x;
    if (g >= E * EDIMC) return;
    int e = g >> 5, c = g & 31;
    int d = dst[e];
    atomicAdd(loopea + (size_t)d * EDIMC + c, ew[g]);
    if (c == 0) atomicAdd(cnt + d, 1);
}

__global__ void loopea_div(float* __restrict__ loopea, const int* __restrict__ cnt, int Nn) {
    int g = blockIdx.x * blockDim.x + threadIdx.x;
    if (g >= Nn * EDIMC) return;
    float c = (float)cnt[g >> 5];
    loopea[g] /= (c < 1.f ? 1.f : c);
}

// ---------------------------------------------------------------------------
// Edge scores + segment max (uint-ordered atomicMax)
// ---------------------------------------------------------------------------
__global__ void gat_score(const int* __restrict__ src, const int* __restrict__ dst,
                          const float* __restrict__ xl, const float* __restrict__ xr,
                          const float* __restrict__ ee, const float* __restrict__ att,
                          float* __restrict__ score, unsigned* __restrict__ mkeys,
                          int E, int Nn) {
    int g = blockIdx.x * blockDim.x + threadIdx.x;
    if (g >= (E + Nn) * HEADSC) return;
    int j = g >> 3, h = g & 7;
    int s = (j < E) ? src[j] : (j - E);
    int d = (j < E) ? dst[j] : (j - E);
    const float* pl = xl + (size_t)s * DIMC + h * 32;
    const float* pr = xr + (size_t)d * DIMC + h * 32;
    const float* pe = ee + (size_t)j * DIMC + h * 32;
    const float* pa = att + h * 32;
    float sc = 0.f;
    #pragma unroll 4
    for (int c = 0; c < 32; ++c) {
        float z = pl[c] + pr[c] + pe[c];
        z = z > 0.f ? z : 0.2f * z;          // leaky_relu(0.2)
        sc += z * pa[c];
    }
    score[g] = sc;
    atomicMax(mkeys + (size_t)d * HEADSC + h, fkey(sc));
}

__global__ void gat_exp(const int* __restrict__ dst, float* __restrict__ score,
                        const unsigned* __restrict__ mkeys, float* __restrict__ den,
                        int E, int Nn) {
    int g = blockIdx.x * blockDim.x + threadIdx.x;
    if (g >= (E + Nn) * HEADSC) return;
    int j = g >> 3, h = g & 7;
    int d = (j < E) ? dst[j] : (j - E);
    float m = fkeyinv(mkeys[(size_t)d * HEADSC + h]);
    float ex = expf(score[g] - m);
    score[g] = ex;
    atomicAdd(den + (size_t)d * HEADSC + h, ex);
}

// one block (256 thr) per edge: out[dst] += alpha * xl[src]
__global__ void gat_aggr(const int* __restrict__ src, const int* __restrict__ dst,
                         const float* __restrict__ xl, const float* __restrict__ ex,
                         const float* __restrict__ den, float* __restrict__ out,
                         int E, int Nn) {
    int j = blockIdx.x;
    int t = threadIdx.x;              // channel 0..255
    int s = (j < E) ? src[j] : (j - E);
    int d = (j < E) ? dst[j] : (j - E);
    int h = t >> 5;
    float alpha = ex[(size_t)j * HEADSC + h] / den[(size_t)d * HEADSC + h];
    atomicAdd(out + (size_t)d * DIMC + t, alpha * xl[(size_t)s * DIMC + t]);
}

// ---------------------------------------------------------------------------
// BatchNorm (training batch stats, biased var) + leaky_relu(0.01) + residual
// ---------------------------------------------------------------------------
__global__ __launch_bounds__(256)
void bn_stats(const float* __restrict__ x, int rows, int C,
              float* __restrict__ mu, float* __restrict__ inv) {
    int c = blockIdx.x, t = threadIdx.x;
    float s = 0.f, s2 = 0.f;
    for (int r = t; r < rows; r += 256) {
        float v = x[(size_t)r * C + c];
        s += v; s2 += v * v;
    }
    __shared__ float sh[256], sh2[256];
    sh[t] = s; sh2[t] = s2;
    __syncthreads();
    for (int o = 128; o > 0; o >>= 1) {
        if (t < o) { sh[t] += sh[t + o]; sh2[t] += sh2[t + o]; }
        __syncthreads();
    }
    if (t == 0) {
        float m = sh[0] / rows;
        float var = sh2[0] / rows - m * m;
        mu[c] = m;
        inv[c] = rsqrtf(var + 1e-5f);
    }
}

__global__ void bn_apply_res(const float* __restrict__ x, const float* __restrict__ res,
                             const float* __restrict__ g, const float* __restrict__ b,
                             const float* __restrict__ mu, const float* __restrict__ inv,
                             float* __restrict__ out, int total, int C) {
    int i = blockIdx.x * blockDim.x + threadIdx.x;
    if (i >= total) return;
    int c = i % C;
    float v = (x[i] - mu[c]) * inv[c] * g[c] + b[c];
    v = v > 0.f ? v : 0.01f * v;              // leaky_relu(0.01)
    out[i] = res[i] + v;
}

// ---------------------------------------------------------------------------
extern "C" void kernel_launch(void* const* d_in, const int* in_sizes, int n_in,
                              void* d_out, int out_size, void* d_ws, size_t ws_size,
                              hipStream_t stream) {
    const float* nf  = (const float*)d_in[0];
    const int*   ei  = (const int*)d_in[1];
    const float* ew  = (const float*)d_in[2];
    const float* g1_Wl = (const float*)d_in[3];
    const float* g1_Wr = (const float*)d_in[4];
    const float* g1_We = (const float*)d_in[5];
    const float* g1_att= (const float*)d_in[6];
    const float* g2_Wl = (const float*)d_in[8];
    const float* g2_Wr = (const float*)d_in[9];
    const float* g2_We = (const float*)d_in[10];
    const float* g2_att= (const float*)d_in[11];
    const float* n1_g = (const float*)d_in[13];
    const float* n1_b = (const float*)d_in[14];
    const float* n2_g = (const float*)d_in[15];
    const float* n2_b = (const float*)d_in[16];
    const float* n3_g = (const float*)d_in[17];
    const float* n3_b = (const float*)d_in[18];
    const float* ffW1 = (const float*)d_in[19];
    const float* ffb1 = (const float*)d_in[20];
    const float* ffW2 = (const float*)d_in[21];
    // g*_b and ff_b2 cancel exactly through BatchNorm mean subtraction.

    const int Nn = in_sizes[0] / DIMC;
    const int E  = in_sizes[1] / 2;
    const int EA = E + Nn;
    const int* src = ei;
    const int* dst = ei + E;

    // ---- workspace carve-out ----
    float* w = (float*)d_ws;
    size_t off = 0;
    auto alloc = [&](size_t n) { float* p = w + off; off += (n + 63) & ~(size_t)63; return p; };
    float*    xl     = alloc((size_t)Nn * DIMC);
    float*    xr     = alloc((size_t)Nn * DIMC);
    float*    gat    = alloc((size_t)Nn * DIMC);     // also reused as ff2
    float*    hbuf   = alloc((size_t)Nn * DIMC);
    float*    loopea = alloc((size_t)Nn * EDIMC);
    int*      cnt    = (int*)alloc((size_t)Nn);
    unsigned* mkeys  = (unsigned*)alloc((size_t)Nn * HEADSC);
    float*    den    = alloc((size_t)Nn * HEADSC);
    float*    mu     = alloc(DIMC);
    float*    inv    = alloc(DIMC);
    float*    sc     = alloc((size_t)EA * HEADSC);
    float*    ff     = alloc((size_t)Nn * DFFC);
    float*    ee     = alloc((size_t)EA * DIMC);     // biggest last

    // ---- self-loop edge-attr mean (shared by both GAT blocks) ----
    hipMemsetAsync(cnt, 0, (size_t)Nn * 4, stream);
    hipMemsetAsync(loopea, 0, (size_t)Nn * EDIMC * 4, stream);
    loopea_accum<<<(E * EDIMC + 255) / 256, 256, 0, stream>>>(dst, ew, loopea, cnt, E);
    loopea_div<<<(Nn * EDIMC + 255) / 256, 256, 0, stream>>>(loopea, cnt, Nn);

    auto gat_block = [&](const float* x, const float* Wl, const float* Wr,
                         const float* We, const float* att,
                         const float* ng, const float* nb, float* hout) {
        dim3 gn((Nn + 63) / 64, DIMC / 64);
        gemm_wmma_bf16<<<gn, 128, 0, stream>>>(x, Wl, nullptr, xl, Nn, DIMC, DIMC, 0);
        gemm_wmma_bf16<<<gn, 128, 0, stream>>>(x, Wr, nullptr, xr, Nn, DIMC, DIMC, 0);
        dim3 ge((E + 63) / 64, DIMC / 64);
        gemm_wmma_bf16<<<ge, 128, 0, stream>>>(ew, We, nullptr, ee, E, DIMC, EDIMC, 0);
        gemm_wmma_bf16<<<gn, 128, 0, stream>>>(loopea, We, nullptr, ee + (size_t)E * DIMC,
                                               Nn, DIMC, EDIMC, 0);
        hipMemsetAsync(mkeys, 0, (size_t)Nn * HEADSC * 4, stream);
        hipMemsetAsync(den, 0, (size_t)Nn * HEADSC * 4, stream);
        int tot = EA * HEADSC;
        gat_score<<<(tot + 255) / 256, 256, 0, stream>>>(src, dst, xl, xr, ee, att,
                                                         sc, mkeys, E, Nn);
        gat_exp<<<(tot + 255) / 256, 256, 0, stream>>>(dst, sc, mkeys, den, E, Nn);
        hipMemsetAsync(gat, 0, (size_t)Nn * DIMC * 4, stream);
        gat_aggr<<<EA, 256, 0, stream>>>(src, dst, xl, sc, den, gat, E, Nn);
        bn_stats<<<DIMC, 256, 0, stream>>>(gat, Nn, DIMC, mu, inv);
        bn_apply_res<<<(Nn * DIMC + 255) / 256, 256, 0, stream>>>(gat, x, ng, nb, mu, inv,
                                                                  hout, Nn * DIMC, DIMC);
    };

    gat_block(nf, g1_Wl, g1_Wr, g1_We, g1_att, n1_g, n1_b, hbuf);
    gat_block(hbuf, g2_Wl, g2_Wr, g2_We, g2_att, n2_g, n2_b, hbuf);

    // ---- FFN: relu(h@W1+b1)@W2, BN(+ff_b2 cancels), lrelu, residual -> d_out ----
    dim3 gf1((Nn + 63) / 64, DFFC / 64);
    gemm_wmma_bf16<<<gf1, 128, 0, stream>>>(hbuf, ffW1, ffb1, ff, Nn, DFFC, DIMC, 1);
    dim3 gf2((Nn + 63) / 64, DIMC / 64);
    gemm_wmma_bf16<<<gf2, 128, 0, stream>>>(ff, ffW2, nullptr, gat, Nn, DIMC, DFFC, 0);
    bn_stats<<<DIMC, 256, 0, stream>>>(gat, Nn, DIMC, mu, inv);
    bn_apply_res<<<(Nn * DIMC + 255) / 256, 256, 0, stream>>>(gat, hbuf, n3_g, n3_b, mu, inv,
                                                              (float*)d_out, Nn * DIMC, DIMC);
}
